// DeepCausalModel_49967649521943
// MI455X (gfx1250) — matrix operand order repaired and checked by reading, not compiled, not run on hardware
//
#include <hip/hip_runtime.h>

// ---------------------------------------------------------------------------
// CDNA5 (gfx1250) wave32 WMMA GEMM chain for DeepCausalModel.
// All heavy math goes through v_wmma_f32_16x16x32_bf16.
// Block tile 128x128, 8 waves in 4(M) x 2(N) grid, each wave 32x64
// (2x4 WMMA fragments -> 8 v_wmma per K-step).
// Global->register->LDS software pipeline: loads for slab kt+1 are in flight
// while slab kt computes; double-buffered LDS.
// ---------------------------------------------------------------------------

typedef __attribute__((ext_vector_type(16))) __bf16 v16bf;
typedef __attribute__((ext_vector_type(8)))  float  v8f;

union FragU {
    v16bf v;
    uint4 q[2];
};

__device__ __forceinline__ unsigned short f2bf(float f) {
    unsigned u = __float_as_uint(f);
    u += 0x7FFFu + ((u >> 16) & 1u);     // round-to-nearest-even
    return (unsigned short)(u >> 16);
}

// ---- fragment loaders (layouts per CDNA5 ISA 7.12.2) ----------------------
// A 16x32 bf16: lane holds M=lane&15; K = kb..kb+7 and kb+16..kb+23, kb=8*(lane>>4)
__device__ __forceinline__ v16bf load_fragA(const unsigned short* base, int ld, int lane) {
    const unsigned short* p = base + (lane & 15) * ld + ((lane >> 4) << 3);
    FragU f;
    f.q[0] = *(const uint4*)(p);
    f.q[1] = *(const uint4*)(p + 16);
    return f.v;
}
// B 32x16 bf16 (stored N-major, K-contiguous in LDS): lane holds N=lane&15;
// K = kb..kb+7 and kb+16..kb+23, kb=8*(lane>>4).
__device__ __forceinline__ v16bf load_fragB(const unsigned short* base, int ld, int lane) {
    const unsigned short* p = base + (lane & 15) * ld + ((lane >> 4) << 3);
    FragU f;
    f.q[0] = *(const uint4*)(p);
    f.q[1] = *(const uint4*)(p + 16);
    return f.v;
}

constexpr int BM = 128, BN = 128, BK = 32;
constexpr int LDA_S = 40;  // padded rows (80B): bank-conflict relief, 16B-aligned
constexpr int LDB_S = 40;

// ---------------------------------------------------------------------------
// C[M,Nout] = ReLU(A[M,K] @ W[K,Nout] + bias); A f32 (layer 0) or bf16.
// Dual-format output: optional f32 and/or bf16 destinations.
// ---------------------------------------------------------------------------
template <bool A_IS_F32>
__global__ __launch_bounds__(256)
void gemm_bias_relu(const void* __restrict__ Ain,
                    const float* __restrict__ W,
                    const float* __restrict__ bias,
                    float* __restrict__ Cf32,
                    unsigned short* __restrict__ Cbf,
                    int M, int K, int Nout) {
    __shared__ alignas(16) unsigned short sA[2][BM * LDA_S];
    __shared__ alignas(16) unsigned short sB[2][BN * LDB_S];

    const int tid  = threadIdx.x;
    const int lane = tid & 31;
    const int wid  = tid >> 5;
    const int waveM = wid & 3;   // 0..3 -> 32-row slice
    const int waveN = wid >> 2;  // 0..1 -> 64-col slice

    const int m0 = blockIdx.y * BM;
    const int n0 = blockIdx.x * BN;
    const int nk = K >> 5;

    const float*          Af = (const float*)Ain;
    const unsigned short* Ab = (const unsigned short*)Ain;

    // per-thread staging coordinates
    const int ar  = A_IS_F32 ? (tid >> 3) : (tid >> 2);
    const int ac  = A_IS_F32 ? ((tid & 7) * 4) : ((tid & 3) * 8);
    const int bn4 = (tid & 31) * 4;    // 0..124 step 4
    const int bkr = tid >> 5;          // 0..7

    // register staging buffers (held across a full K-iteration)
    float4 rA[4];
    uint4  rAb[2];
    float4 rB[4];

    // issue global loads for K-slab kt (no wait here)
    auto gload = [&](int kt) {
        const int k0 = kt * BK;
        if (A_IS_F32) {
#pragma unroll
            for (int p = 0; p < 4; ++p)
                rA[p] = *(const float4*)(Af + (size_t)(m0 + ar + p * 32) * K + k0 + ac);
        } else {
#pragma unroll
            for (int p = 0; p < 2; ++p)
                rAb[p] = *(const uint4*)(Ab + (size_t)(m0 + ar + p * 64) * K + k0 + ac);
        }
#pragma unroll
        for (int i = 0; i < 4; ++i)
            rB[i] = *(const float4*)(W + (size_t)(k0 + bkr + 8 * i) * Nout + n0 + bn4);
    };

    // convert staged registers -> LDS buffer `buf`
    auto commit = [&](int buf) {
        if (A_IS_F32) {
#pragma unroll
            for (int p = 0; p < 4; ++p) {
                unsigned short* d = &sA[buf][(ar + p * 32) * LDA_S + ac];
                d[0] = f2bf(rA[p].x); d[1] = f2bf(rA[p].y);
                d[2] = f2bf(rA[p].z); d[3] = f2bf(rA[p].w);
            }
        } else {
#pragma unroll
            for (int p = 0; p < 2; ++p)
                *(uint4*)(&sA[buf][(ar + p * 64) * LDA_S + ac]) = rAb[p];
        }
#pragma unroll
        for (int i = 0; i < 4; ++i) {
            const int k = bkr + 8 * i;
            sB[buf][(bn4 + 0) * LDB_S + k] = f2bf(rB[i].x);
            sB[buf][(bn4 + 1) * LDB_S + k] = f2bf(rB[i].y);
            sB[buf][(bn4 + 2) * LDB_S + k] = f2bf(rB[i].z);
            sB[buf][(bn4 + 3) * LDB_S + k] = f2bf(rB[i].w);
        }
    };

    v8f acc[2][4] = {};

    gload(0);
    commit(0);
    if (nk > 1) gload(1);

    for (int kt = 0; kt < nk; ++kt) {
        __syncthreads();   // LDS[kt&1] visible to all waves

        const int buf = kt & 1;
        const unsigned short* aBase = &sA[buf][(waveM * 32) * LDA_S];
        const unsigned short* bBase = &sB[buf][(waveN * 64) * LDB_S];

        v16bf aF[2], bF[4];
#pragma unroll
        for (int i = 0; i < 2; ++i) aF[i] = load_fragA(aBase + i * 16 * LDA_S, LDA_S, lane);
#pragma unroll
        for (int j = 0; j < 4; ++j) bF[j] = load_fragB(bBase + j * 16 * LDB_S, LDB_S, lane);

#pragma unroll
        for (int i = 0; i < 2; ++i)
#pragma unroll
            for (int j = 0; j < 4; ++j)
                acc[i][j] = __builtin_amdgcn_wmma_f32_16x16x32_bf16(
                    false, aF[i], false, bF[j], (short)0, acc[i][j], false, false);

        if (kt + 1 < nk) {
            commit((kt + 1) & 1);          // regs loaded one iteration ago
            if (kt + 2 < nk) gload(kt + 2); // next loads go in flight now
        }
    }

    // ---- epilogue: bias + ReLU, dual-format store ----
    const int rbase = m0 + waveM * 32 + ((lane >> 4) << 3);
    const int cbase = n0 + waveN * 64 + (lane & 15);
#pragma unroll
    for (int i = 0; i < 2; ++i) {
#pragma unroll
        for (int j = 0; j < 4; ++j) {
            const int col = cbase + j * 16;
            const float b = bias[col];
#pragma unroll
            for (int e = 0; e < 8; ++e) {
                float v = acc[i][j][e] + b;
                v = v > 0.0f ? v : 0.0f;
                const size_t idx = (size_t)(rbase + i * 16 + e) * Nout + col;
                if (Cf32) Cf32[idx] = v;
                if (Cbf)  Cbf[idx]  = f2bf(v);
            }
        }
    }
}

// ---------------------------------------------------------------------------
// Head: y[n] = dot(G2[t[n]][n][:512], Wo[t[n]]) + bo[t[n]];  t_out[n] = 1.0
// ---------------------------------------------------------------------------
__global__ __launch_bounds__(256)
void head_kernel(const unsigned short* __restrict__ G2,
                 const int* __restrict__ treat,
                 const float* __restrict__ Wo,
                 const float* __restrict__ bo,
                 float* __restrict__ y,
                 float* __restrict__ tout,
                 int N) {
    const int n = blockIdx.x * 256 + threadIdx.x;
    if (n >= N) return;
    const int t = treat[n];
    const uint4* g = (const uint4*)(G2 + ((size_t)t * N + (size_t)n) * 512);
    const float* w = Wo + t * 512;
    float s = bo[t];
#pragma unroll 4
    for (int q = 0; q < 64; ++q) {
        const uint4 v = g[q];
        const unsigned vv[4] = {v.x, v.y, v.z, v.w};
#pragma unroll
        for (int e = 0; e < 4; ++e) {
            const float lo = __uint_as_float(vv[e] << 16);
            const float hi = __uint_as_float(vv[e] & 0xFFFF0000u);
            s += lo * w[q * 8 + e * 2] + hi * w[q * 8 + e * 2 + 1];
        }
    }
    y[n] = s;
    tout[n] = 1.0f;
}

// ---------------------------------------------------------------------------
extern "C" void kernel_launch(void* const* d_in, const int* in_sizes, int n_in,
                              void* d_out, int out_size, void* d_ws, size_t ws_size,
                              hipStream_t stream) {
    (void)in_sizes; (void)n_in; (void)out_size; (void)ws_size;

    const int N = 65536;
    const float* x   = (const float*)d_in[0];
    const int*   trt = (const int*)d_in[1];
    const float* Wx0 = (const float*)d_in[2];  const float* bx0 = (const float*)d_in[3];
    const float* Wx1 = (const float*)d_in[4];  const float* bx1 = (const float*)d_in[5];
    const float* Wx2 = (const float*)d_in[6];  const float* bx2 = (const float*)d_in[7];
    const float* Wy0 = (const float*)d_in[8];  const float* by0 = (const float*)d_in[9];
    const float* Wy1 = (const float*)d_in[10]; const float* by1 = (const float*)d_in[11];
    const float* Wo  = (const float*)d_in[12]; const float* bo  = (const float*)d_in[13];
    // Wt (d_in[14]) / bt (d_in[15]) unused: softmax over a size-1 axis == 1.0

    float* out   = (float*)d_out;
    float* y     = out;                       // [N,1]
    float* xemb  = out + N;                   // [N,512]
    float* tout  = out + N + (size_t)N * 512; // [N,1]

    char* ws = (char*)d_ws;
    unsigned short* H1 = (unsigned short*)(ws);                       // [N,2048] bf16 (256MiB)
    unsigned short* H2 = (unsigned short*)(ws + (size_t)268435456);   // [N,2048] bf16 (256MiB)
    unsigned short* XE = (unsigned short*)(ws + (size_t)536870912);   // [N,512]  bf16 (64MiB)
    unsigned short* G1 = H1;                                          // [2,N,1024] bf16 (reuse)
    unsigned short* G2 = H2;                                          // [2,N,512]  bf16 (reuse)

    const dim3 blk(256);
    const unsigned gy = N / BM; // 512

    // trunk
    gemm_bias_relu<true ><<<dim3(2048 / BN, gy), blk, 0, stream>>>(x,  Wx0, bx0, nullptr, H1, N, 512,  2048);
    gemm_bias_relu<false><<<dim3(2048 / BN, gy), blk, 0, stream>>>(H1, Wx1, bx1, nullptr, H2, N, 2048, 2048);
    gemm_bias_relu<false><<<dim3(512  / BN, gy), blk, 0, stream>>>(H2, Wx2, bx2, xemb,    XE, N, 2048, 512);

    // experts (dense over T=2)
    for (int t = 0; t < 2; ++t) {
        gemm_bias_relu<false><<<dim3(1024 / BN, gy), blk, 0, stream>>>(
            XE, Wy0 + (size_t)t * 512 * 1024, by0 + t * 1024,
            nullptr, G1 + (size_t)t * N * 1024, N, 512, 1024);
    }
    for (int t = 0; t < 2; ++t) {
        gemm_bias_relu<false><<<dim3(512 / BN, gy), blk, 0, stream>>>(
            G1 + (size_t)t * N * 1024, Wy1 + (size_t)t * 1024 * 512, by1 + t * 512,
            nullptr, G2 + (size_t)t * N * 512, N, 1024, 512);
    }

    // gather + output head + propensity(=1.0)
    head_kernel<<<dim3(N / 256), blk, 0, stream>>>(G2, trt, Wo, bo, y, tout, N);
}
